// POLY2Model_12421045420619
// MI455X (gfx1250) — matrix operand order repaired and checked by reading, not compiled, not run on hardware
//
#include <hip/hip_runtime.h>

#define F        1024
#define BATCH    16384
#define BM       32           // batch rows per workgroup
#define BN       256          // N columns per tile iteration
#define KB       64           // K rows of W staged in LDS per tile
#define NTILES   (F / BN)     // 4
#define KBLOCKS  (F / KB)     // 16
#define TILES    (NTILES * KBLOCKS)  // 64
#define XSTRIDE  (F + 8)      // padded Xs row stride (halves)
#define WSTRIDE  (KB + 8)     // padded Ws row stride (halves)

typedef __attribute__((ext_vector_type(16))) _Float16 v16h;
typedef __attribute__((ext_vector_type(8)))  _Float16 v8h;
typedef __attribute__((ext_vector_type(4)))  _Float16 v4h;
typedef __attribute__((ext_vector_type(8)))  float    v8f;
typedef __attribute__((ext_vector_type(4)))  float    v4f;

union F16Frag { v16h v; v8h h[2]; };

// Kernel 1: build dense transposed f16 weight matrix Wt[n][k] = W[k][n],
// W strictly upper triangular from flat w in triu(k=1) row-major order:
// p(i,j) = i*F - i*(i+1)/2 + (j - i - 1), valid for j > i.
__global__ __launch_bounds__(256) void poly2_build_wt(const float* __restrict__ w,
                                                      _Float16* __restrict__ wt) {
  int idx = blockIdx.x * 256 + threadIdx.x;   // over F*F elements
  int n = idx >> 10;          // column of W (row of Wt)
  int k = idx & (F - 1);      // row of W
  _Float16 v = (_Float16)0.f;
  if (k < n) {
    int p = k * F - (k * (k + 1)) / 2 + (n - k - 1);
    v = (_Float16)w[p];
  }
  wt[idx] = v;
}

// Kernel 2: Y = X_block (32xF) * W (FxF) via f16 WMMA with async-LDS W tiles,
// fused with out_b = sigmoid(sum_n Y[b,n] * x[b,n]).
__global__ __launch_bounds__(256) void poly2_gemm(const float* __restrict__ x,
                                                  const _Float16* __restrict__ wt,
                                                  float* __restrict__ out) {
  __shared__ __align__(16) _Float16 Xs[BM * XSTRIDE];        // ~64.5 KB persistent f16 X
  __shared__ __align__(16) _Float16 Wsb[2][BN * WSTRIDE];    // 2 x 36 KB W tiles (async dest)
  __shared__ float outAcc[BM];

  const int tid  = threadIdx.x;
  const int lane = tid & 31;
  const int wave = tid >> 5;          // 0..7
  const int wm   = wave & 1;          // 2 waves in M
  const int wn   = wave >> 1;         // 4 waves in N, 64 cols each
  const int row0 = blockIdx.x * BM;

  if (tid < BM) outAcc[tid] = 0.f;

  // ---- Phase 1: stream X block fp32 -> f16 LDS (coalesced 16B loads) ----
#pragma unroll
  for (int i = 0; i < (BM * F / 4) / 256; ++i) {   // 32 iterations
    int idx4 = tid + i * 256;
    int r  = idx4 >> 8;                            // F/4 = 256 float4 per row
    int c4 = idx4 & 255;
    v4f xv = *(const v4f*)(x + (size_t)(row0 + r) * F + c4 * 4);
    v4h hv;
    hv[0] = (_Float16)xv[0]; hv[1] = (_Float16)xv[1];
    hv[2] = (_Float16)xv[2]; hv[3] = (_Float16)xv[3];
    *(v4h*)(Xs + r * XSTRIDE + c4 * 4) = hv;
  }

  // Async prefetch of one 256x64 W tile (32 KB) into LDS buffer `buf`.
  // 256 threads x 8 chunks x 16B; 8 consecutive lanes cover one 128B row (coalesced).
  auto prefetch = [&](int tt, int buf) {
    const int n0 = (tt >> 4) * BN;                 // KBLOCKS == 16
    const int k0 = (tt & (KBLOCKS - 1)) * KB;
    const int rn = tid >> 3;                       // 0..31
    const int rk = (tid & 7) * 8;                  // halves
    const _Float16* g = wt + (size_t)(n0 + rn) * F + (k0 + rk);
    unsigned int l = (unsigned int)(unsigned long long)(&Wsb[buf][rn * WSTRIDE + rk]);
#pragma unroll
    for (int i = 0; i < 8; ++i) {
      asm volatile("global_load_async_to_lds_b128 %0, %1, off"
                   :: "v"(l), "v"(g) : "memory");
      g += 32 * F;                 // next 32-row group in global W
      l += 32 * WSTRIDE * 2;       // next 32-row group in LDS (bytes)
    }
  };

  // Fragment addressing (wave32 WMMA layouts per CDNA5 ISA):
  const int mlane = lane & 15;
  const int lhi   = lane >> 4;               // 0 or 1
  const int arow  = wm * 16 + mlane;         // A: M = lane&15
  const int akb   = lhi * 8;                 // A: k-base 0/8 (plus +16 chunk)
  const int bkb   = lhi * 8;                 // B: k-base 0/8 (plus +16 chunk)

  float dotp[8];
#pragma unroll
  for (int v = 0; v < 8; ++v) dotp[v] = 0.f;
  v8f acc[4];
  const v8f vzero = {};
#pragma unroll
  for (int j = 0; j < 4; ++j) acc[j] = vzero;

  prefetch(0, 0);

  for (int t = 0; t < TILES; ++t) {
    // own async loads landed; barrier => tile t visible, tile t-1 compute done
    asm volatile("s_wait_asynccnt 0x0" ::: "memory");
    __syncthreads();

    const _Float16* wsc = Wsb[t & 1];
    const int k0 = (t & (KBLOCKS - 1)) * KB;

    // ---- Issue ALL fragment loads for this tile first (20 x ds_load_b128),
    //      so WMMAs below can drain with partial dscnt waits. ----
    F16Frag A0, A1, B[8];
    {
      const _Float16* ap = Xs + arow * XSTRIDE + (k0 + akb);
      A0.h[0] = *(const v8h*)(ap);          // ks=0  K[akb..akb+7], K[akb+16..+23]
      A0.h[1] = *(const v8h*)(ap + 16);
      A1.h[0] = *(const v8h*)(ap + 32);     // ks=32
      A1.h[1] = *(const v8h*)(ap + 48);
      const _Float16* bp0 = wsc + (wn * 64 + mlane) * WSTRIDE + bkb;
#pragma unroll
      for (int j = 0; j < 4; ++j) {
        const _Float16* bp = bp0 + j * 16 * WSTRIDE;
        B[j].h[0]     = *(const v8h*)(bp);          // ks=0  K[bkb..bkb+7], K[bkb+16..+23]
        B[j].h[1]     = *(const v8h*)(bp + 16);
        B[j + 4].h[0] = *(const v8h*)(bp + 32);     // ks=32
        B[j + 4].h[1] = *(const v8h*)(bp + 48);
      }
    }

    // overlap: issue next tile's async loads while fragments are in flight
    if (t + 1 < TILES) prefetch(t + 1, (t + 1) & 1);

    // ---- 8 back-to-back WMMAs (4 independent accumulator chains) ----
#pragma unroll
    for (int j = 0; j < 4; ++j)
      acc[j] = __builtin_amdgcn_wmma_f32_16x16x32_f16(false, A0.v, false, B[j].v,
                                                      (short)0, acc[j], false, false);
#pragma unroll
    for (int j = 0; j < 4; ++j)
      acc[j] = __builtin_amdgcn_wmma_f32_16x16x32_f16(false, A1.v, false, B[j + 4].v,
                                                      (short)0, acc[j], false, false);

    // end of an N-tile: fold C into per-row dot partials, reset accumulators
    if ((t & (KBLOCKS - 1)) == (KBLOCKS - 1)) {
      const int n0 = (t >> 4) * BN;
#pragma unroll
      for (int v = 0; v < 8; ++v) {
        int r = wm * 16 + (lhi ? (v + 8) : v);
#pragma unroll
        for (int j = 0; j < 4; ++j) {
          int c = n0 + wn * 64 + j * 16 + mlane;
          dotp[v] += acc[j][v] * (float)Xs[r * XSTRIDE + c];
        }
      }
#pragma unroll
      for (int j = 0; j < 4; ++j) acc[j] = vzero;
    }
  }

  // cross-lane / cross-wave reduction via LDS float atomics
#pragma unroll
  for (int v = 0; v < 8; ++v) {
    int r = wm * 16 + (lhi ? (v + 8) : v);
    atomicAdd(&outAcc[r], dotp[v]);
  }
  __syncthreads();

  if (tid < BM) {
    float s = outAcc[tid];
    out[row0 + tid] = 1.f / (1.f + __expf(-s));
  }
}

extern "C" void kernel_launch(void* const* d_in, const int* in_sizes, int n_in,
                              void* d_out, int out_size, void* d_ws, size_t ws_size,
                              hipStream_t stream) {
  (void)in_sizes; (void)n_in; (void)out_size; (void)ws_size;
  const float* x = (const float*)d_in[0];   // [BATCH, F] fp32
  const float* w = (const float*)d_in[1];   // [CROSS_DIMS] fp32
  float* out = (float*)d_out;               // [BATCH] fp32
  _Float16* wt = (_Float16*)d_ws;           // 2 MB dense transposed f16 W

  poly2_build_wt<<<(F * F) / 256, 256, 0, stream>>>(w, wt);
  poly2_gemm<<<BATCH / BM, 256, 0, stream>>>(x, wt, out);
}
